// LocalSelfAttention_68564857914104
// MI455X (gfx1250) — matrix operand order repaired and checked by reading, not compiled, not run on hardware
//
#include <hip/hip_runtime.h>
#include <hip/hip_bf16.h>

// ---------------------------------------------------------------------------
// LocalSelfAttention for MI455X (gfx1250, wave32, WMMA bf16 path)
// B=4, S=2048, D_IN=512, UNITS=64, WIN=32 -> W=65, L=132064
// ---------------------------------------------------------------------------

typedef __attribute__((ext_vector_type(16))) __bf16 v16bf;
typedef __attribute__((ext_vector_type(8)))  float  v8f;

#define S_LEN   2048
#define D_IN    512
#define UNITS   64
#define WIN     32
#define WPAD    80   // 65 window rows padded to 5 x 16
#define KPAD    96   // K-dim padding for the weights @ V GEMM (3 x 32)

// fragment-order weight buffer geometry (per projection):
//   frag f = kt*4 + nv  (kt: 16 K-tiles, nv: 4 N-tiles), 32 lanes, 16 bf16 each
#define WFRAGS  ((D_IN / 32) * (UNITS / 16))     // 64 frags
#define WSWZ_ELEMS ((size_t)WFRAGS * 32 * 16)    // 32768 bf16 per projection

// --- WMMA fragment loaders (contiguous k order k = 16*(lane/16) + elem, same for A and B) -------------------

// A-matrix 16x32 bf16, row-major source with element stride `stride`.
// lane 0-15 : row m=lane,   elems 0..7 -> K 0..7,   elems 8..15 -> K 8..15
// lane 16-31: row m=lane-16,elems 0..7 -> K 16..23, elems 8..15 -> K 24..31
__device__ inline v16bf load_frag_a(const __bf16* base, int stride) {
    int lane = threadIdx.x & 31;
    int m    = lane & 15;
    int k0   = (lane >> 4) ? 16 : 0;
    int k1   = (lane >> 4) ? 24 : 8;
    const __bf16* row = base + m * stride;
    v16bf a;
#pragma unroll
    for (int e = 0; e < 8; ++e) a[e]     = row[k0 + e];   // merges -> ds_load_b128
#pragma unroll
    for (int e = 0; e < 8; ++e) a[8 + e] = row[k1 + e];   // merges -> ds_load_b128
    return a;
}

// B-matrix 32x16 bf16: element (k,n) at base[k*strideK + n*strideN].
// lane 0-15: col n=lane, K 0..15 ; lane 16-31: col n=lane-16, K 16..31
// Use strideK==1 call sites so the 16 elements merge into b128 loads.
__device__ inline v16bf load_frag_b(const __bf16* base, int strideK, int strideN) {
    int lane = threadIdx.x & 31;
    int n    = lane & 15;
    int kb   = (lane >> 4) << 4;
    v16bf b;
#pragma unroll
    for (int e = 0; e < 16; ++e) b[e] = base[(kb + e) * strideK + n * strideN];
    return b;
}

__device__ inline v8f wmma_bf16(v16bf a, v16bf b, v8f c) {
    return __builtin_amdgcn_wmma_f32_16x16x32_bf16(
        /*neg_a=*/false, a, /*neg_b=*/false, b,
        /*c_mod=*/(short)0, c, /*reuse_a=*/false, /*reuse_b=*/false);
}

// ---------------------------------------------------------------------------
// Weight pre-swizzle: fp32 (512x64) -> bf16 in WMMA B-fragment order, so the
// projection GEMM loads each fragment as one contiguous 32B run per lane.
// One thread = one (proj, frag, lane) slot (3 * 64 * 32 = 6144 threads).
// ---------------------------------------------------------------------------
__global__ __launch_bounds__(256)
void lsa_wswz_kernel(const float* __restrict__ Wq, const float* __restrict__ Wk,
                     const float* __restrict__ Wv, __bf16* __restrict__ Wsw) {
    int gid = blockIdx.x * 256 + threadIdx.x;
    int p   = gid / (WFRAGS * 32);
    int rem = gid % (WFRAGS * 32);
    int f   = rem >> 5;
    int l   = rem & 31;
    int kt  = f >> 2, nv = f & 3;
    int n   = l & 15, kb = (l >> 4) << 4;
    const float* W = (p == 0) ? Wq : (p == 1) ? Wk : Wv;
    __bf16* dst = Wsw + (size_t)p * WSWZ_ELEMS + ((size_t)f * 32 + l) * 16;
#pragma unroll
    for (int e = 0; e < 16; ++e)
        dst[e] = (__bf16)W[(size_t)(kt * 32 + kb + e) * UNITS + nv * 16 + n];
}

// ---------------------------------------------------------------------------
// Projection: Y(bf16, ROWSx64) = X(f32, ROWSx512) @ W + bias
// One block = 16 rows; 4 waves, one 16-col N-tile per wave; 16 K-steps.
// B fragments come pre-swizzled from Wsw (2x global_load_b128 per step).
// ---------------------------------------------------------------------------
__global__ __launch_bounds__(128)
void lsa_proj_kernel(const float* __restrict__ X, const __bf16* __restrict__ Wsw,
                     const float* __restrict__ bias, __bf16* __restrict__ Y) {
    __shared__ __align__(16) __bf16 Xs[16 * D_IN];

    const int tid     = threadIdx.x;
    const size_t rowb = (size_t)blockIdx.x * 16;

    // Stage 16x512 fp32 tile -> bf16 LDS (rows of X are contiguous).
    const float* src = X + rowb * D_IN;
    for (int i = tid * 4; i < 16 * D_IN; i += 128 * 4) {
        float4 v = *(const float4*)(src + i);
        Xs[i + 0] = (__bf16)v.x; Xs[i + 1] = (__bf16)v.y;
        Xs[i + 2] = (__bf16)v.z; Xs[i + 3] = (__bf16)v.w;
    }
    __syncthreads();

    const int wave = tid >> 5;
    const int lane = tid & 31;

    v8f c = {};
#pragma unroll 4
    for (int kt = 0; kt < D_IN / 32; ++kt) {
        v16bf a = load_frag_a(Xs + kt * 32, D_IN);
        v16bf b = *(const v16bf*)(Wsw + ((size_t)(kt * 4 + wave) * 32 + lane) * 16);
        c = wmma_bf16(a, b, c);
    }

    const int n  = lane & 15;
    const int mb = (lane >> 4) * 8;
    const float bv = bias[wave * 16 + n];
#pragma unroll
    for (int r = 0; r < 8; ++r) {
        Y[(rowb + mb + r) * UNITS + wave * 16 + n] = (__bf16)(c[r] + bv);
    }
}

// ---------------------------------------------------------------------------
// Attention: one block per (s, b). Window rows [start,end) of Q/K/V.
//  GEMM1: Sc(80x80,f32) = Qw(80x64) @ Kw^T        (25 tiles, 2 K-steps)
//  softmax rows (masked to len) -> Wt(80x96,bf16, zero padded)
//  GEMM2: out(80x64,f32) = Wt @ Vt^T(96x64)       (20 tiles, 3 K-steps)
// V is staged TRANSPOSED in LDS (Vt[d][u], 64x96) so GEMM2 B-fragment
// elements are contiguous along K and merge into ds_load_b128.
// LDS reuse: Qs+Ks region is recycled as Wt after GEMM1.
// ---------------------------------------------------------------------------
__global__ __launch_bounds__(256)
void lsa_attn_kernel(const __bf16* __restrict__ Qb, const __bf16* __restrict__ Kb,
                     const __bf16* __restrict__ Vb, float* __restrict__ out,
                     int Ltot) {
    __shared__ __align__(16) unsigned char smem[WPAD*UNITS*2*2 + UNITS*KPAD*2 + WPAD*WPAD*4];
    __bf16* Qs = (__bf16*)smem;                                   // 80x64 bf16
    __bf16* Ks = (__bf16*)(smem + WPAD*UNITS*2);                  // 80x64 bf16
    __bf16* Vt = (__bf16*)(smem + 2*WPAD*UNITS*2);                // 64x96 bf16 (transposed)
    float*  Sc = (float*) (smem + 2*WPAD*UNITS*2 + UNITS*KPAD*2); // 80x80 f32
    __bf16* Wt = (__bf16*)smem;                                   // 80x96 bf16 (reuse)

    const int s   = blockIdx.x;
    const int b   = blockIdx.y;
    const int tid = threadIdx.x;

    const int start = max(s - WIN, 0);
    const int end   = min(s + WIN + 1, S_LEN);
    const int len   = end - start;

    // Closed-form ragged offset: offs[s] = 65*s - head - tail
    const int head  = (s <= WIN) ? (WIN * s - (s * (s - 1)) / 2) : (WIN * (WIN + 1)) / 2;
    const int tt    = s - (S_LEN - WIN);                 // s - 2016
    const int tail  = (tt > 0) ? (tt * (tt + 1)) / 2 : 0;
    const long offs = (long)(2 * WIN + 1) * s - head - tail;

    // --- Stage Q/K window (clamped rows) and zero-padded transposed V -----
    const __bf16* Qg = Qb + (size_t)b * S_LEN * UNITS;
    const __bf16* Kg = Kb + (size_t)b * S_LEN * UNITS;
    const __bf16* Vg = Vb + (size_t)b * S_LEN * UNITS;

    for (int i = tid; i < WPAD * (UNITS / 2); i += 256) {      // 2 bf16 / iter
        int r = i >> 5, cp = i & 31;
        int g = min(start + r, S_LEN - 1);
        ((unsigned int*)Qs)[i] = ((const unsigned int*)(Qg + (size_t)g * UNITS))[cp];
        ((unsigned int*)Ks)[i] = ((const unsigned int*)(Kg + (size_t)g * UNITS))[cp];
    }
    for (int i = tid; i < KPAD * (UNITS / 2); i += 256) {      // u-major walk
        int u = i >> 5, dp = i & 31;
        union { unsigned int u32; __bf16 h[2]; } cv;
        cv.u32 = 0u;
        if (u < len)
            cv.u32 = ((const unsigned int*)(Vg + (size_t)(start + u) * UNITS))[dp];
        Vt[(2 * dp + 0) * KPAD + u] = cv.h[0];
        Vt[(2 * dp + 1) * KPAD + u] = cv.h[1];
    }
    __syncthreads();

    const int wave = tid >> 5;
    const int lane = tid & 31;
    const int n    = lane & 15;
    const int mb   = (lane >> 4) * 8;

    // --- GEMM1: scores = Qw @ Kw^T * scale --------------------------------
    const float scale = 0.125f;                       // 1/sqrt(64)
    for (int t = wave; t < 25; t += 8) {
        int mt = t / 5, nt = t % 5;
        v8f c = {};
#pragma unroll
        for (int kt = 0; kt < UNITS / 32; ++kt) {
            v16bf a  = load_frag_a(Qs + mt * 16 * UNITS + kt * 32, UNITS);
            // B[k][u] = Kw[u][k]: contiguous along k -> b128 merges
            v16bf bm = load_frag_b(Ks + nt * 16 * UNITS + kt * 32, 1, UNITS);
            c = wmma_bf16(a, bm, c);
        }
#pragma unroll
        for (int r = 0; r < 8; ++r)
            Sc[(mt * 16 + mb + r) * WPAD + nt * 16 + n] = c[r] * scale;
    }
    __syncthreads();

    // --- Masked row softmax -> bf16 weights (zero padded to 96) -----------
    if (tid < WPAD) {
        float* row = Sc + tid * WPAD;
        float m = -3.402823466e38f;
        for (int u = 0; u < len; ++u) m = fmaxf(m, row[u]);
        float sum = 0.f;
        for (int u = 0; u < len; ++u) { float e = __expf(row[u] - m); row[u] = e; sum += e; }
        float inv = 1.f / sum;
        __bf16* wrow = Wt + tid * KPAD;
        for (int u = 0; u < KPAD; ++u)
            wrow[u] = (__bf16)((u < len) ? row[u] * inv : 0.f);
    }
    __syncthreads();

    // --- GEMM2: out = weights @ Vw, ragged store --------------------------
    float* outb = out + (size_t)b * Ltot * UNITS;
    for (int t = wave; t < 20; t += 8) {
        int mt = t / 4, nt = t % 4;
        v8f c = {};
#pragma unroll
        for (int kt = 0; kt < KPAD / 32; ++kt) {
            v16bf a  = load_frag_a(Wt + mt * 16 * KPAD + kt * 32, KPAD);
            // B[k=u][n=d] = Vt[d][u]: contiguous along k -> b128 merges
            v16bf bm = load_frag_b(Vt + nt * 16 * KPAD + kt * 32, 1, KPAD);
            c = wmma_bf16(a, bm, c);
        }
#pragma unroll
        for (int r = 0; r < 8; ++r) {
            int w = mt * 16 + mb + r;
            if (w < len)
                outb[(offs + w) * UNITS + nt * 16 + n] = c[r];
        }
    }
}

// ---------------------------------------------------------------------------
extern "C" void kernel_launch(void* const* d_in, const int* in_sizes, int n_in,
                              void* d_out, int out_size, void* d_ws, size_t ws_size,
                              hipStream_t stream) {
    (void)in_sizes; (void)n_in; (void)ws_size;
    const float* query = (const float*)d_in[0];
    const float* value = (const float*)d_in[1];
    const float* Wq    = (const float*)d_in[2];
    const float* bq    = (const float*)d_in[3];
    const float* Wk    = (const float*)d_in[4];
    const float* bk    = (const float*)d_in[5];
    const float* Wv    = (const float*)d_in[6];
    const float* bv    = (const float*)d_in[7];

    const size_t nQKV = (size_t)4 * S_LEN * UNITS;     // per-tensor bf16 elems
    __bf16* Qb  = (__bf16*)d_ws;
    __bf16* Kb  = Qb + nQKV;
    __bf16* Vb  = Kb + nQKV;
    __bf16* Wsw = Vb + nQKV;                           // 3 * 32768 bf16

    // Pre-swizzle weights into fragment order (24 blocks x 256 = 6144 thr).
    lsa_wswz_kernel<<<(3 * WFRAGS * 32) / 256, 256, 0, stream>>>(Wq, Wk, Wv, Wsw);

    const int rowTiles = (4 * S_LEN) / 16;             // 512 blocks
    lsa_proj_kernel<<<rowTiles, 128, 0, stream>>>(query, Wsw + 0 * WSWZ_ELEMS, bq, Qb);
    lsa_proj_kernel<<<rowTiles, 128, 0, stream>>>(value, Wsw + 1 * WSWZ_ELEMS, bk, Kb);
    lsa_proj_kernel<<<rowTiles, 128, 0, stream>>>(value, Wsw + 2 * WSWZ_ELEMS, bv, Vb);

    const int Ltot = out_size / (4 * UNITS);           // 132064
    dim3 grid(S_LEN, 4);
    lsa_attn_kernel<<<grid, 256, 0, stream>>>(Qb, Kb, Vb, (float*)d_out, Ltot);
}
